// ResidualAPPNPWrapper_7267084664913
// MI455X (gfx1250) — matrix operand order repaired and checked by reading, not compiled, not run on hardware
//
#include <hip/hip_runtime.h>

// ---------------------------------------------------------------------------
// Residual-MLP + APPNP for MI455X (gfx1250, wave32, WMMA).
// MLP: 4 bf16-WMMA GEMM passes with fused bias / LayerNorm / ReLU epilogues.
// APPNP: 10 steps of L2-resident gather + f32-atomic scatter-add.
// ---------------------------------------------------------------------------

#define NNODES 100000
#define IN_C   512
#define HID    256
#define OUT_C  128
#define KSTEPS 10

typedef __attribute__((ext_vector_type(16))) __bf16 v16bf;
typedef __attribute__((ext_vector_type(8)))  float  v8f;

// float -> bf16 bits with round-to-nearest-even (pure integer ops)
static __device__ __forceinline__ unsigned short bfbits(float f) {
    unsigned u = __builtin_bit_cast(unsigned, f);
    unsigned r = (u + 0x7FFFu + ((u >> 16) & 1u)) >> 16;
    return (unsigned short)r;
}

// ---------------------------------------------------------------------------
// Pack a row-major fp32 K x N weight matrix into a wave32 WMMA-B
// fragment layout for v_wmma_f32_16x16x32_bf16 (contiguous k order, same as the A loads):
//   lane l, element e:  K = ktile*32 + 16*(l>=16) + e,  N = ntile*16 + (l&15)
// stored as [ktile][ntile][lane][16] bf16 so each lane loads 32 contiguous B.
// ---------------------------------------------------------------------------
__global__ void pack_wmma_b_kernel(const float* __restrict__ W,
                                   unsigned short* __restrict__ out,
                                   int K, int Ncols) {
    int nTiles = Ncols >> 4;
    int kTiles = K >> 5;
    int idx = blockIdx.x * blockDim.x + threadIdx.x;
    int total = kTiles * nTiles * 32;
    if (idx >= total) return;
    int lane  = idx & 31;
    int tile  = idx >> 5;
    int ntile = tile % nTiles;
    int kt    = tile / nTiles;
    int col   = ntile * 16 + (lane & 15);
    int kbase = kt * 32 + ((lane >> 4) ? 16 : 0);
    unsigned short* o = out + (size_t)idx * 16;
#pragma unroll
    for (int e = 0; e < 16; ++e)
        o[e] = bfbits(W[(size_t)(kbase + e) * Ncols + col]);
}

// ---------------------------------------------------------------------------
// WMMA GEMM: C[M x NCOLS] = A[M x K] @ B + bias, optional LayerNorm+ReLU.
// 256 threads = 8 waves; block computes 16 rows x NCOLS. Wave w owns
// n-tiles [w*TPW, (w+1)*TPW). TPW and NCOLS are compile-time so the
// accumulators live in fixed even-aligned VGPR ranges (no v_movrel indexing).
// ---------------------------------------------------------------------------
template <int TPW, int DOLN>
__global__ __launch_bounds__(256)
void gemm_wmma_kernel(const float* __restrict__ A,
                      const unsigned short* __restrict__ Bp,
                      const float* __restrict__ bias,
                      const float* __restrict__ gamma,
                      const float* __restrict__ beta,
                      float* __restrict__ C,
                      int K) {
    constexpr int NCOLS  = 128 * TPW;
    constexpr int NTILES = NCOLS / 16;
    __shared__ float lds[16 * NCOLS];

    const int lane = threadIdx.x & 31;
    const int wv   = threadIdx.x >> 5;
    const int hi   = lane >> 4;               // 0: lanes 0-15, 1: lanes 16-31
    const int m0   = blockIdx.x << 4;
    const int row  = m0 + (lane & 15);

    const float* Arow = A + (size_t)row * K + (hi ? 16 : 0);

    v8f acc[TPW];
#pragma unroll
    for (int t = 0; t < TPW; ++t)
#pragma unroll
        for (int i = 0; i < 8; ++i) acc[t][i] = 0.0f;

    const int kTiles = K >> 5;
    // per-wave B base: n-tile index wv*TPW, stride between k-tiles = NTILES*512
    const unsigned short* bptr =
        Bp + ((size_t)(wv * TPW) * 32 + lane) * 16;

    for (int kt = 0; kt < kTiles; ++kt) {
        const float* ap = Arow + kt * 32;
        float4 c0a = *(const float4*)(ap);
        float4 c0b = *(const float4*)(ap + 4);
        float4 c1a = *(const float4*)(ap + 8);
        float4 c1b = *(const float4*)(ap + 12);
        union { v16bf v; unsigned short u[16]; } a;
        a.u[0]  = bfbits(c0a.x); a.u[1]  = bfbits(c0a.y);
        a.u[2]  = bfbits(c0a.z); a.u[3]  = bfbits(c0a.w);
        a.u[4]  = bfbits(c0b.x); a.u[5]  = bfbits(c0b.y);
        a.u[6]  = bfbits(c0b.z); a.u[7]  = bfbits(c0b.w);
        a.u[8]  = bfbits(c1a.x); a.u[9]  = bfbits(c1a.y);
        a.u[10] = bfbits(c1a.z); a.u[11] = bfbits(c1a.w);
        a.u[12] = bfbits(c1b.x); a.u[13] = bfbits(c1b.y);
        a.u[14] = bfbits(c1b.z); a.u[15] = bfbits(c1b.w);

        const unsigned short* bk = bptr + (size_t)kt * NTILES * 512;
#pragma unroll
        for (int t = 0; t < TPW; ++t) {
            const v16bf b = *(const v16bf*)(bk + (size_t)t * 512);
            acc[t] = __builtin_amdgcn_wmma_f32_16x16x32_bf16(
                false, a.v, false, b, (short)0, acc[t], false, false);
        }
    }

    // Spill accumulators to LDS: VGPR v of lane l holds (M = v + 8*hi, N = col)
#pragma unroll
    for (int t = 0; t < TPW; ++t) {
        int col = (wv * TPW + t) * 16 + (lane & 15);
#pragma unroll
        for (int v = 0; v < 8; ++v)
            lds[(hi * 8 + v) * NCOLS + col] = acc[t][v];
    }
    __syncthreads();

    // Epilogue: 16 threads per row (threads r*16..r*16+15 are one half-wave).
    const int r  = threadIdx.x >> 4;
    const int c0 = threadIdx.x & 15;
    constexpr int PER = NCOLS / 16;
    float* orow = C + (size_t)(m0 + r) * NCOLS;

    if constexpr (!DOLN) {
#pragma unroll
        for (int j = 0; j < PER; ++j) {
            int c = c0 + (j << 4);
            orow[c] = lds[r * NCOLS + c] + bias[c];
        }
    } else {
        float s = 0.0f, s2 = 0.0f;
#pragma unroll
        for (int j = 0; j < PER; ++j) {
            int c = c0 + (j << 4);
            float v = lds[r * NCOLS + c] + bias[c];
            s += v; s2 += v * v;
        }
        // butterfly within the 16-lane half of the wave
#pragma unroll
        for (int o = 8; o >= 1; o >>= 1) {
            s  += __shfl_xor(s, o);
            s2 += __shfl_xor(s2, o);
        }
        constexpr float inv = 1.0f / (float)NCOLS;
        float mu  = s * inv;
        float var = s2 * inv - mu * mu;
        float rs  = __frsqrt_rn(var + 1e-5f);
#pragma unroll
        for (int j = 0; j < PER; ++j) {
            int c = c0 + (j << 4);
            float v = lds[r * NCOLS + c] + bias[c];
            float y = (v - mu) * rs * gamma[c] + beta[c];
            orow[c] = fmaxf(y, 0.0f);
        }
    }
}

// ---------------------------------------------------------------------------
// APPNP helpers
// ---------------------------------------------------------------------------
__global__ void deg_init_kernel(float* __restrict__ deg, int n) {
    int i = blockIdx.x * blockDim.x + threadIdx.x;
    if (i < n) deg[i] = 1.0f;                 // self loop
}

__global__ void deg_accum_kernel(const int* __restrict__ dst,
                                 float* __restrict__ deg, int e) {
    int i = blockIdx.x * blockDim.x + threadIdx.x;
    if (i < e) atomicAdd(&deg[dst[i]], 1.0f);
}

__global__ void deg_fin_kernel(float* __restrict__ deg, int n) {
    int i = blockIdx.x * blockDim.x + threadIdx.x;
    if (i < n) deg[i] = __frsqrt_rn(deg[i]);  // deg >= 1 always
}

// per-edge weight with (1-alpha) folded in
__global__ void edge_norm_kernel(const int* __restrict__ src,
                                 const int* __restrict__ dst,
                                 const float* __restrict__ dinv,
                                 float* __restrict__ enorm, int e) {
    int i = blockIdx.x * blockDim.x + threadIdx.x;
    if (i < e) enorm[i] = 0.9f * dinv[src[i]] * dinv[dst[i]];
}

// next = alpha*h0 + (1-alpha)*dinv[i]^2*cur  (teleport + self-loop term)
__global__ void prop_init_kernel(const float* __restrict__ h0,
                                 const float* __restrict__ cur,
                                 const float* __restrict__ dinv,
                                 float* __restrict__ next, int n) {
    int idx = blockIdx.x * blockDim.x + threadIdx.x;
    if (idx >= n * OUT_C) return;
    float di = dinv[idx >> 7];
    next[idx] = 0.1f * h0[idx] + 0.9f * di * di * cur[idx];
}

// one wave per edge; lane handles 4 channels (float4 gather, 4 f32 atomics).
// cur/next (51 MB each) are L2-resident on MI455X (192 MB L2), so the
// atomics resolve in L2 at far above HBM rate.
__global__ __launch_bounds__(256)
void prop_edges_kernel(const int* __restrict__ src,
                       const int* __restrict__ dst,
                       const float* __restrict__ enorm,
                       const float* __restrict__ cur,
                       float* __restrict__ next, int nedges) {
    int gid  = blockIdx.x * blockDim.x + threadIdx.x;
    int e    = gid >> 5;
    if (e >= nedges) return;
    int lane = gid & 31;
    int s = src[e], d = dst[e];
    float w = enorm[e];
    const float4 v = *(const float4*)(cur + (size_t)s * OUT_C + lane * 4);
    float* o = next + (size_t)d * OUT_C + lane * 4;
    atomicAdd(o + 0, w * v.x);
    atomicAdd(o + 1, w * v.y);
    atomicAdd(o + 2, w * v.z);
    atomicAdd(o + 3, w * v.w);
}

// ---------------------------------------------------------------------------
extern "C" void kernel_launch(void* const* d_in, const int* in_sizes, int n_in,
                              void* d_out, int out_size, void* d_ws, size_t ws_size,
                              hipStream_t stream) {
    const float* x     = (const float*)d_in[0];
    const int*   ei    = (const int*)  d_in[1];
    const float* W_in  = (const float*)d_in[2];
    const float* b_in  = (const float*)d_in[3];
    const float* W1    = (const float*)d_in[4];
    const float* b1    = (const float*)d_in[5];
    const float* g1    = (const float*)d_in[6];
    const float* be1   = (const float*)d_in[7];
    const float* W2    = (const float*)d_in[8];
    const float* b2    = (const float*)d_in[9];
    const float* g2    = (const float*)d_in[10];
    const float* be2   = (const float*)d_in[11];
    const float* W_out = (const float*)d_in[12];
    const float* b_out = (const float*)d_in[13];

    const int E = in_sizes[1] / 2;
    const int* src = ei;
    const int* dst = ei + E;

    // workspace carve
    char* p = (char*)d_ws;
    auto take = [&](size_t bytes) {
        char* r = p;
        p += (bytes + 255) & ~(size_t)255;
        return r;
    };
    float*          bufA  = (float*)take((size_t)NNODES * HID * 4);
    float*          bufB  = (float*)take((size_t)NNODES * HID * 4);
    float*          dinv  = (float*)take((size_t)NNODES * 4);
    float*          enorm = (float*)take((size_t)E * 4);
    unsigned short* WpIn  = (unsigned short*)take((size_t)IN_C * HID * 2);
    unsigned short* Wp1   = (unsigned short*)take((size_t)HID * HID * 2);
    unsigned short* Wp2   = (unsigned short*)take((size_t)HID * HID * 2);
    unsigned short* WpOut = (unsigned short*)take((size_t)HID * OUT_C * 2);

    // --- pack weights into WMMA-B fragment layout (tiny) ---
    {
        int t0 = (IN_C / 32) * (HID / 16) * 32;
        int t1 = (HID / 32) * (HID / 16) * 32;
        int t3 = (HID / 32) * (OUT_C / 16) * 32;
        pack_wmma_b_kernel<<<(t0 + 255) / 256, 256, 0, stream>>>(W_in, WpIn, IN_C, HID);
        pack_wmma_b_kernel<<<(t1 + 255) / 256, 256, 0, stream>>>(W1, Wp1, HID, HID);
        pack_wmma_b_kernel<<<(t1 + 255) / 256, 256, 0, stream>>>(W2, Wp2, HID, HID);
        pack_wmma_b_kernel<<<(t3 + 255) / 256, 256, 0, stream>>>(W_out, WpOut, HID, OUT_C);
    }

    // --- MLP: 4 WMMA GEMM passes (RES_ALPHA=1 => block = relu(LN(h@W+b))) ---
    const int gBlocks = NNODES / 16;   // 6250
    gemm_wmma_kernel<2, 0><<<gBlocks, 256, 0, stream>>>(x,    WpIn,  b_in,  g1, be1, bufA, IN_C);
    gemm_wmma_kernel<2, 1><<<gBlocks, 256, 0, stream>>>(bufA, Wp1,   b1,    g1, be1, bufB, HID);
    gemm_wmma_kernel<2, 1><<<gBlocks, 256, 0, stream>>>(bufB, Wp2,   b2,    g2, be2, bufA, HID);
    gemm_wmma_kernel<1, 0><<<gBlocks, 256, 0, stream>>>(bufA, WpOut, b_out, g1, be1, bufB, HID);
    float* h0 = bufB;                  // N x 128 fp32

    // --- degree normalization ---
    deg_init_kernel <<<(NNODES + 255) / 256, 256, 0, stream>>>(dinv, NNODES);
    deg_accum_kernel<<<(E + 255) / 256,      256, 0, stream>>>(dst, dinv, E);
    deg_fin_kernel  <<<(NNODES + 255) / 256, 256, 0, stream>>>(dinv, NNODES);
    edge_norm_kernel<<<(E + 255) / 256,      256, 0, stream>>>(src, dst, dinv, enorm, E);

    // --- 10 propagation steps; ping-pong bufA <-> d_out, last step -> d_out ---
    float* P1 = bufA;                  // h3 no longer needed
    const float* cur = h0;
    const int initBlocks = (NNODES * OUT_C + 255) / 256;     // 50000
    const int edgeBlocks = (int)(((size_t)E * 32 + 255) / 256);  // 200000
    for (int k = 0; k < KSTEPS; ++k) {
        float* next = (k & 1) ? (float*)d_out : P1;          // k=9 -> d_out
        prop_init_kernel <<<initBlocks, 256, 0, stream>>>(h0, cur, dinv, next, NNODES);
        prop_edges_kernel<<<edgeBlocks, 256, 0, stream>>>(src, dst, enorm, cur, next, E);
        cur = next;
    }
}
